// RecursiveTransformer_65008624992822
// MI455X (gfx1250) — matrix operand order repaired and checked by reading, not compiled, not run on hardware
//
#include <hip/hip_runtime.h>
#include <hip/hip_bf16.h>

#define H_  12
#define D_  768
#define DH_ 64
#define FF_ 3072
#define N_  8
#define L_  1024
#define VL_ 512
#define M_  (N_ * L_)   // 8192 rows

typedef __attribute__((ext_vector_type(16))) __bf16 v16bf;
typedef __attribute__((ext_vector_type(8)))  float  v8f;
typedef int v4i_t __attribute__((ext_vector_type(4)));

#if defined(__gfx1250__) && __has_builtin(__builtin_amdgcn_global_load_async_to_lds_b128) && __has_builtin(__builtin_amdgcn_s_wait_asynccnt)
#define USE_ASYNC_LDS 1
typedef __attribute__((address_space(1))) v4i_t* gv4p;
typedef __attribute__((address_space(3))) v4i_t* lv4p;
#else
#define USE_ASYNC_LDS 0
#endif

static __device__ __forceinline__ unsigned short f2bf(float f) {
  return __builtin_bit_cast(unsigned short, static_cast<__bf16>(f));  // native RNE cvt
}
// Build a 16x32 A / 32x16 B bf16 WMMA fragment from two aligned 16B chunks.
static __device__ __forceinline__ v16bf ld_frag(const unsigned short* p0,
                                                const unsigned short* p1) {
  struct U4x2 { uint4 a, b; } t;
  t.a = *(const uint4*)p0;
  t.b = *(const uint4*)p1;
  return __builtin_bit_cast(v16bf, t);
}

// ------------------------------------------------------------------
// f32 -> bf16 convert
// ------------------------------------------------------------------
__global__ __launch_bounds__(256) void cvt_kernel(const float* __restrict__ in,
                                                  unsigned short* __restrict__ out,
                                                  int n) {
  int i = blockIdx.x * 256 + threadIdx.x;
  if (i < n) out[i] = f2bf(in[i]);
}

// ------------------------------------------------------------------
// bf16 GEMM: C = A(MxK) * B(KxN) + bias [+resid fp32][gelu]
// block tile 128x128x32, 8 waves (2x4), wave tile 64x32 (4x2 WMMA tiles)
// A tile staged via async global->LDS (ASYNCcnt) when available.
// B tile stored transposed in LDS so fragment reads are 2x ds_load_b128.
// TRANSV: write bf16 output transposed as (b, h, dh, L) with b128 stores.
// ------------------------------------------------------------------
template <bool RESID, bool GELU, bool OUTF, bool OUTB, bool TRANSV>
__global__ __launch_bounds__(256) void gemm_kernel(
    const unsigned short* __restrict__ A,
    const unsigned short* __restrict__ B,
    const float* __restrict__ bias,
    const float* __restrict__ resid,
    float* __restrict__ outF,
    unsigned short* __restrict__ outB,
    int M, int Nn, int K)
{
  __shared__ alignas(16) unsigned short As[128 * 32];   // row-major (m, k)
  __shared__ alignas(16) unsigned short Bt[128 * 32];   // transposed (n, k)

  const int tid  = threadIdx.x;
  const int lane = tid & 31;
  const int wid  = tid >> 5;
  const int wm   = wid >> 2;      // 0..1
  const int wn   = wid & 3;       // 0..3
  const int lh   = lane >> 4;     // lane half
  const int ln   = lane & 15;
  const int bm   = blockIdx.y * 128;
  const int bn   = blockIdx.x * 128;

  const int cg = tid & 15;        // B: column group (8 cols)
  const int rp = tid >> 4;        // B: row pair (k = 2rp, 2rp+1)

  v8f acc[4][2];
#pragma unroll
  for (int mt = 0; mt < 4; ++mt)
#pragma unroll
    for (int nt = 0; nt < 2; ++nt)
#pragma unroll
      for (int i = 0; i < 8; ++i) acc[mt][nt][i] = 0.0f;

  for (int kt = 0; kt < K; kt += 32) {
    // ---- stage A tile (128x32 row-major), 16B chunks ----
    {
      int idx = tid;
#pragma unroll
      for (int rep = 0; rep < 2; ++rep, idx += 256) {
        int row = idx >> 2, q = idx & 3;
#if USE_ASYNC_LDS
        __builtin_amdgcn_global_load_async_to_lds_b128(
            (gv4p)(A + (size_t)(bm + row) * K + kt + q * 8),
            (lv4p)(As + idx * 8), 0, 0);
#else
        ((uint4*)As)[idx] = ((const uint4*)(A + (size_t)(bm + row) * K + kt))[q];
#endif
      }
    }
    // ---- stage B tile transposed: load 2 rows x 8 cols, pack pairs ----
    {
      union U4 { uint4 v; unsigned short s[8]; } ra, rb;
      ra.v = ((const uint4*)(B + (size_t)(kt + 2 * rp) * Nn + bn))[cg];
      rb.v = ((const uint4*)(B + (size_t)(kt + 2 * rp + 1) * Nn + bn))[cg];
      unsigned int* bt32 = (unsigned int*)Bt;
#pragma unroll
      for (int j = 0; j < 8; ++j)
        bt32[(cg * 8 + j) * 16 + rp] =
            (unsigned int)ra.s[j] | ((unsigned int)rb.s[j] << 16);
    }
#if USE_ASYNC_LDS
    __builtin_amdgcn_s_wait_asynccnt(0);
#endif
    __syncthreads();

    v16bf af[4], bfr[2];
#pragma unroll
    for (int mt = 0; mt < 4; ++mt) {
      const int m = wm * 64 + mt * 16 + ln;
      af[mt] = ld_frag(&As[m * 32 + lh * 8], &As[m * 32 + 16 + lh * 8]);
    }
#pragma unroll
    for (int nt = 0; nt < 2; ++nt) {
      const int n = wn * 32 + nt * 16 + ln;
      bfr[nt] = ld_frag(&Bt[n * 32 + lh * 8], &Bt[n * 32 + 16 + lh * 8]);
    }
#pragma unroll
    for (int mt = 0; mt < 4; ++mt)
#pragma unroll
      for (int nt = 0; nt < 2; ++nt)
        acc[mt][nt] = __builtin_amdgcn_wmma_f32_16x16x32_bf16(
            false, af[mt], false, bfr[nt], (short)0, acc[mt][nt], false, false);
    __syncthreads();
  }

  // ---- epilogue (C layout: row = r + 8*lh, col = ln within 16x16) ----
#pragma unroll
  for (int mt = 0; mt < 4; ++mt) {
#pragma unroll
    for (int nt = 0; nt < 2; ++nt) {
      const int col = bn + wn * 32 + nt * 16 + ln;
      const float bcol = bias[col];
      if (TRANSV) {
        // V transposed store: rows row0..row0+7 are consecutive tokens
        const int row0 = bm + wm * 64 + mt * 16 + 8 * lh;
        const int bidx = row0 >> 10;          // / L_
        const int l0 = row0 & (L_ - 1);
        const int hh = col >> 6, dh = col & 63;
        union PK { uint4 v; unsigned short s[8]; } pk;
#pragma unroll
        for (int r = 0; r < 8; ++r) pk.s[r] = f2bf(acc[mt][nt][r] + bcol);
        *(uint4*)(outB + ((size_t)(bidx * H_ + hh) * DH_ + dh) * L_ + l0) = pk.v;
      } else {
#pragma unroll
        for (int r = 0; r < 8; ++r) {
          const int row = bm + wm * 64 + mt * 16 + r + 8 * lh;
          float v = acc[mt][nt][r] + bcol;
          if (RESID) v += resid[(size_t)row * Nn + col];
          if (GELU)  v = 0.5f * v * (1.0f + erff(v * 0.70710678118654752f));
          if (OUTF)  outF[(size_t)row * Nn + col] = v;
          if (OUTB)  outB[(size_t)row * Nn + col] = f2bf(v);
        }
      }
    }
  }
}

// ------------------------------------------------------------------
// Flash attention: 1 wave per (b, h, 16-query tile); streams 32-key blocks
// V is pre-transposed to (b, h, dh, L) so all fragment loads are b128.
// Mask values are loaded unconditionally per key block (keys depend only
// on the lane), predicate applied arithmetically -> no EXEC divergence.
// ------------------------------------------------------------------
__global__ __launch_bounds__(32) void attn_kernel(
    const unsigned short* __restrict__ Q,
    const unsigned short* __restrict__ Kb,
    const unsigned short* __restrict__ Vt,
    const int* __restrict__ im,
    unsigned short* __restrict__ ctx)
{
  __shared__ alignas(16) unsigned short Ps[16 * 32];
  const int lane = threadIdx.x;
  const int lh = lane >> 4, ln = lane & 15;
  const int q0 = blockIdx.x * 16;
  const int h  = blockIdx.y;
  const int b  = blockIdx.z;
  const size_t base  = ((size_t)b * L_) * D_ + (size_t)h * DH_;
  const size_t vbase = (size_t)(b * H_ + h) * DH_ * L_;
  const float scale = 0.125f;  // 1/sqrt(64)

  // Q as two A fragments (d = 0..31, 32..63); chunks {lh*8, 16+lh*8}
  v16bf aq[2];
#pragma unroll
  for (int kc = 0; kc < 2; ++kc) {
    const unsigned short* qp = Q + base + (size_t)(q0 + ln) * D_ + kc * 32 + lh * 8;
    aq[kc] = ld_frag(qp, qp + 16);
  }

  v8f o[4];
#pragma unroll
  for (int nt = 0; nt < 4; ++nt)
#pragma unroll
    for (int i = 0; i < 8; ++i) o[nt][i] = 0.0f;
  float mrun[8], lrun[8];
#pragma unroll
  for (int r = 0; r < 8; ++r) { mrun[r] = -3.0e38f; lrun[r] = 0.0f; }

  v8f z;
#pragma unroll
  for (int i = 0; i < 8; ++i) z[i] = 0.0f;

  for (int kbk = 0; kbk < L_; kbk += 32) {
    const int key0 = kbk + ln, key1 = kbk + 16 + ln;
    // unconditional pad-mask loads (keys are r-invariant)
    const float mf0 = (float)im[b * L_ + key0];
    const float mf1 = (float)im[b * L_ + key1];

    // K^T fragments: B(d x key); chunks {lh*8, 16+lh*8} within kc*32
    v16bf bkf[2][2];
#pragma unroll
    for (int nt = 0; nt < 2; ++nt)
#pragma unroll
      for (int kc = 0; kc < 2; ++kc) {
        const unsigned short* kp =
            Kb + base + (size_t)(kbk + nt * 16 + ln) * D_ + kc * 32 + lh * 8;
        bkf[nt][kc] = ld_frag(kp, kp + 16);
      }
    v8f s0 = __builtin_amdgcn_wmma_f32_16x16x32_bf16(false, aq[0], false, bkf[0][0], (short)0, z, false, false);
    s0     = __builtin_amdgcn_wmma_f32_16x16x32_bf16(false, aq[1], false, bkf[0][1], (short)0, s0, false, false);
    v8f s1 = __builtin_amdgcn_wmma_f32_16x16x32_bf16(false, aq[0], false, bkf[1][0], (short)0, z, false, false);
    s1     = __builtin_amdgcn_wmma_f32_16x16x32_bf16(false, aq[1], false, bkf[1][1], (short)0, s1, false, false);

    float corr[8];
#pragma unroll
    for (int r = 0; r < 8; ++r) {
      const int qrow = q0 + r + 8 * lh;
      // shifted mask: visual keys (<512) always visible; causal among text
      const bool a0 = (key0 < VL_) | ((qrow >= VL_) & (key0 <= qrow));
      const bool a1 = (key1 < VL_) | ((qrow >= VL_) & (key1 <= qrow));
      const float m0 = a0 ? mf0 : 0.0f;
      const float m1 = a1 ? mf1 : 0.0f;
      float v0 = s0[r] * scale + (1.0f - m0) * -10000.0f;
      float v1 = s1[r] * scale + (1.0f - m1) * -10000.0f;
      float lm = fmaxf(v0, v1);
#pragma unroll
      for (int xm = 8; xm >= 1; xm >>= 1) lm = fmaxf(lm, __shfl_xor(lm, xm, 16));
      float nm = fmaxf(mrun[r], lm);
      float c  = __expf(mrun[r] - nm);
      mrun[r] = nm;
      corr[r] = c;
      float p0 = __expf(v0 - nm), p1 = __expf(v1 - nm);
      float rs = p0 + p1;
#pragma unroll
      for (int xm = 8; xm >= 1; xm >>= 1) rs += __shfl_xor(rs, xm, 16);
      lrun[r] = lrun[r] * c + rs;
      Ps[(r + 8 * lh) * 32 + ln]      = f2bf(p0);
      Ps[(r + 8 * lh) * 32 + 16 + ln] = f2bf(p1);
    }
#pragma unroll
    for (int nt = 0; nt < 4; ++nt)
#pragma unroll
      for (int r = 0; r < 8; ++r) o[nt][r] *= corr[r];
    __syncthreads();

    // P back as an A fragment (16 queries x 32 keys)
    v16bf ap = ld_frag(&Ps[ln * 32 + lh * 8], &Ps[ln * 32 + 16 + lh * 8]);

    // V fragments from transposed V: chunks key = kbk + {lh*8, 16+lh*8}
#pragma unroll
    for (int nt = 0; nt < 4; ++nt) {
      const unsigned short* vp =
          Vt + vbase + (size_t)(nt * 16 + ln) * L_ + kbk + lh * 8;
      v16bf bvf = ld_frag(vp, vp + 16);
      o[nt] = __builtin_amdgcn_wmma_f32_16x16x32_bf16(false, ap, false, bvf, (short)0, o[nt], false, false);
    }
    __syncthreads();
  }

#pragma unroll
  for (int nt = 0; nt < 4; ++nt)
#pragma unroll
    for (int r = 0; r < 8; ++r) {
      float v = o[nt][r] / lrun[r];
      ctx[base + (size_t)(q0 + r + 8 * lh) * D_ + nt * 16 + ln] = f2bf(v);
    }
}

// ------------------------------------------------------------------
// fp32 LayerNorm over D_=768, one block per row
// ------------------------------------------------------------------
__global__ __launch_bounds__(256) void ln_kernel(
    const float* __restrict__ x, const float* __restrict__ w,
    const float* __restrict__ bb,
    float* __restrict__ outF, unsigned short* __restrict__ outB)
{
  __shared__ float red[256];
  const int row = blockIdx.x;
  const int tid = threadIdx.x;
  const float* xr = x + (size_t)row * D_;
  float s = 0.f;
  for (int i = tid; i < D_; i += 256) s += xr[i];
  red[tid] = s; __syncthreads();
  for (int o = 128; o > 0; o >>= 1) { if (tid < o) red[tid] += red[tid + o]; __syncthreads(); }
  const float mean = red[0] * (1.0f / D_);
  __syncthreads();
  float v = 0.f;
  for (int i = tid; i < D_; i += 256) { float d = xr[i] - mean; v += d * d; }
  red[tid] = v; __syncthreads();
  for (int o = 128; o > 0; o >>= 1) { if (tid < o) red[tid] += red[tid + o]; __syncthreads(); }
  const float inv = rsqrtf(red[0] * (1.0f / D_) + 1e-12f);
  for (int i = tid; i < D_; i += 256) {
    float y = w[i] * (xr[i] - mean) * inv + bb[i];
    if (outF) outF[(size_t)row * D_ + i] = y;
    if (outB) outB[(size_t)row * D_ + i] = f2bf(y);
  }
}

// ------------------------------------------------------------------
extern "C" void kernel_launch(void* const* d_in, const int* in_sizes, int n_in,
                              void* d_out, int out_size, void* d_ws, size_t ws_size,
                              hipStream_t stream) {
  (void)in_sizes; (void)n_in; (void)out_size; (void)ws_size;
  const float* x_in = (const float*)d_in[0];
  const int*   mask = (const int*)d_in[1];
  const float* Wq  = (const float*)d_in[2];
  const float* bq  = (const float*)d_in[3];
  const float* Wk  = (const float*)d_in[4];
  const float* bk  = (const float*)d_in[5];
  const float* Wv  = (const float*)d_in[6];
  const float* bv  = (const float*)d_in[7];
  const float* Wo  = (const float*)d_in[8];
  const float* bo  = (const float*)d_in[9];
  const float* ln1w = (const float*)d_in[10];
  const float* ln1b = (const float*)d_in[11];
  const float* Wi  = (const float*)d_in[12];
  const float* bi  = (const float*)d_in[13];
  const float* Wf  = (const float*)d_in[14];
  const float* bfv = (const float*)d_in[15];
  const float* ln2w = (const float*)d_in[16];
  const float* ln2b = (const float*)d_in[17];

  const size_t NLD  = (size_t)M_ * D_;
  const size_t NLFF = (size_t)M_ * FF_;
  const size_t DD   = (size_t)D_ * D_;
  const size_t DFF  = (size_t)D_ * FF_;

  char* wsb = (char*)d_ws;
  size_t off = 0;
  auto take = [&](size_t bytes) -> void* {
    void* r = wsb + off;
    off = (off + bytes + 255) & ~(size_t)255;
    return r;
  };
  unsigned short* xb     = (unsigned short*)take(NLD * 2);
  unsigned short* wqb    = (unsigned short*)take(DD * 2);
  unsigned short* wkb    = (unsigned short*)take(DD * 2);
  unsigned short* wvb    = (unsigned short*)take(DD * 2);
  unsigned short* wob    = (unsigned short*)take(DD * 2);
  unsigned short* wib    = (unsigned short*)take(DFF * 2);
  unsigned short* wfb    = (unsigned short*)take(DFF * 2);
  unsigned short* qb     = (unsigned short*)take(NLD * 2);
  unsigned short* kbuf   = (unsigned short*)take(NLD * 2);
  unsigned short* vt     = (unsigned short*)take(NLD * 2);   // (b,h,dh,L)
  unsigned short* ctxb   = (unsigned short*)take(NLD * 2);
  unsigned short* attnb  = (unsigned short*)take(NLD * 2);
  unsigned short* interb = (unsigned short*)take(NLFF * 2);
  float* tmpf  = (float*)take(NLD * 4);
  float* attnf = (float*)take(NLD * 4);
  float* xf    = (float*)take(NLD * 4);

  auto cvt = [&](const float* src, unsigned short* dst, int n) {
    cvt_kernel<<<dim3((n + 255) / 256), 256, 0, stream>>>(src, dst, n);
  };

  // weights -> bf16 (once; reused by both weight-shared layers)
  cvt(Wq, wqb, (int)DD);
  cvt(Wk, wkb, (int)DD);
  cvt(Wv, wvb, (int)DD);
  cvt(Wo, wob, (int)DD);
  cvt(Wi, wib, (int)DFF);
  cvt(Wf, wfb, (int)DFF);

  for (int layer = 0; layer < 2; ++layer) {
    const float* xsrc = (layer == 0) ? x_in : xf;
    cvt(xsrc, xb, (int)NLD);
    // Q, K: plain bf16 out
    gemm_kernel<false, false, false, true, false>
        <<<dim3(D_ / 128, M_ / 128), 256, 0, stream>>>(
            xb, wqb, bq, nullptr, nullptr, qb, M_, D_, D_);
    gemm_kernel<false, false, false, true, false>
        <<<dim3(D_ / 128, M_ / 128), 256, 0, stream>>>(
            xb, wkb, bk, nullptr, nullptr, kbuf, M_, D_, D_);
    // V: transposed bf16 out (b, h, dh, L)
    gemm_kernel<false, false, false, true, true>
        <<<dim3(D_ / 128, M_ / 128), 256, 0, stream>>>(
            xb, wvb, bv, nullptr, nullptr, vt, M_, D_, D_);
    attn_kernel<<<dim3(L_ / 16, H_, N_), 32, 0, stream>>>(qb, kbuf, vt, mask, ctxb);
    // Wo: f32 out with residual
    gemm_kernel<true, false, true, false, false>
        <<<dim3(D_ / 128, M_ / 128), 256, 0, stream>>>(
            ctxb, wob, bo, xsrc, tmpf, nullptr, M_, D_, D_);
    ln_kernel<<<dim3(M_), 256, 0, stream>>>(tmpf, ln1w, ln1b, attnf, attnb);
    // Wi: bf16 out with GELU
    gemm_kernel<false, true, false, true, false>
        <<<dim3(FF_ / 128, M_ / 128), 256, 0, stream>>>(
            attnb, wib, bi, nullptr, nullptr, interb, M_, FF_, D_);
    // Wf: f32 out with residual
    gemm_kernel<true, false, true, false, false>
        <<<dim3(D_ / 128, M_ / 128), 256, 0, stream>>>(
            interb, wfb, bfv, attnf, tmpf, nullptr, M_, D_, FF_);
    float* lnout = (layer == 1) ? (float*)d_out : xf;
    ln_kernel<<<dim3(M_), 256, 0, stream>>>(tmpf, ln2w, ln2b, lnout, nullptr);
  }
}
